// GNNPolicy_82094004896465
// MI455X (gfx1250) — matrix operand order repaired and checked by reading, not compiled, not run on hardware
//
#include <hip/hip_runtime.h>
#include <hip/hip_bf16.h>
#include <math.h>

typedef __attribute__((ext_vector_type(16))) _Float16 v16h;
typedef __attribute__((ext_vector_type(8)))  float    v8f;

#define N_NODES 1536

// Low 32 bits of a flat shared-memory pointer = LDS byte offset (flat LDS
// addresses are {shared_aperture_hi32, lds_offset32}).
__device__ __forceinline__ unsigned lds_off(const void* p) {
    return (unsigned)(size_t)p;
}

// ---------------------------------------------------------------------------
// One-shot weight swizzle: B[K,N] f32 (row-major) -> fragment-native f16.
//   halfIdx = (((ntile*(K/32) + kstep)*32 + lane)*16 + h)
//   <- B[(kstep*32 + 8*(lane>>4) + (h&7) + 16*(h>>3)) * ldb + (ntile*16 + (lane&15))]
// ---------------------------------------------------------------------------
__global__ __launch_bounds__(256)
void swizzle_B_kernel(const float* __restrict__ B, int ldb, int K,
                      _Float16* __restrict__ Bf, int total)
{
    const int idx = blockIdx.x * blockDim.x + threadIdx.x;
    if (idx >= total) return;
    const int h      = idx & 15;
    const int lane   = (idx >> 4) & 31;
    const int rest   = idx >> 9;
    const int ksteps = K >> 5;
    const int kstep  = rest % ksteps;
    const int ntile  = rest / ksteps;
    const int k = kstep * 32 + ((lane >> 4) << 3) + (h & 7) + ((h >> 3) << 4);
    const int n = ntile * 16 + (lane & 15);
    Bf[idx] = (_Float16)B[k * ldb + n];
}

// ---------------------------------------------------------------------------
// Encoder layer 1: h1 = relu(x @ W1 + b1), K=4 -> plain VALU (a 4-deep K
// wastes 7/8 of a 16x16x32 WMMA). One thread per (row, channel).
// ---------------------------------------------------------------------------
__global__ __launch_bounds__(256)
void enc1_kernel(const float* __restrict__ x,   // [N,4]
                 const float* __restrict__ W1,  // [4,64]
                 const float* __restrict__ b1,  // [64]
                 float* __restrict__ h1)        // [N,64]
{
    const int idx = blockIdx.x * blockDim.x + threadIdx.x;   // i*64 + c
    const int i = idx >> 6, c = idx & 63;
    const float4 xr = *reinterpret_cast<const float4*>(x + i * 4);
    float v = xr.x * W1[0 * 64 + c] + xr.y * W1[1 * 64 + c]
            + xr.z * W1[2 * 64 + c] + xr.w * W1[3 * 64 + c] + b1[c];
    h1[idx] = fmaxf(v, 0.0f);
}

// ---------------------------------------------------------------------------
// Tiled GEMM: C[M,N] = act(A[M,K] @ B + bias), f16 WMMA, f32 accumulate.
// Compile-time K (multiple of 32) -> fully unrolled, zero guards. One wave
// computes a 16x64 tile; A fragment = 4 unconditional float4 loads, B
// fragment = one aligned 32-byte v16h load from the pre-swizzled buffer.
// ---------------------------------------------------------------------------
template <int K>
__global__ __launch_bounds__(32)
void gemm_wmma_kernel(const float* __restrict__ A, int lda,
                      const _Float16* __restrict__ Bf,
                      const float* __restrict__ bias,
                      float* __restrict__ C, int ldc,
                      int do_relu)
{
    constexpr int KSTEPS = K / 32;
    const int lane = threadIdx.x;            // 0..31 (wave32)
    const int row0 = blockIdx.x * 16;
    const int lm   = lane & 15;
    const int hi   = lane >> 4;

    const float* __restrict__ Arow = A + (row0 + lm) * lda;
    const v16h* __restrict__ Bfv = reinterpret_cast<const v16h*>(Bf);

    v8f acc[4] = {};

    #pragma unroll
    for (int ks = 0; ks < KSTEPS; ++ks) {
        const int base = ks * 32 + hi * 8;
        const float4 a0 = *reinterpret_cast<const float4*>(Arow + base);
        const float4 a1 = *reinterpret_cast<const float4*>(Arow + base + 4);
        const float4 a2 = *reinterpret_cast<const float4*>(Arow + base + 16);
        const float4 a3 = *reinterpret_cast<const float4*>(Arow + base + 20);
        v16h af;
        af[0]  = (_Float16)a0.x; af[1]  = (_Float16)a0.y;
        af[2]  = (_Float16)a0.z; af[3]  = (_Float16)a0.w;
        af[4]  = (_Float16)a1.x; af[5]  = (_Float16)a1.y;
        af[6]  = (_Float16)a1.z; af[7]  = (_Float16)a1.w;
        af[8]  = (_Float16)a2.x; af[9]  = (_Float16)a2.y;
        af[10] = (_Float16)a2.z; af[11] = (_Float16)a2.w;
        af[12] = (_Float16)a3.x; af[13] = (_Float16)a3.y;
        af[14] = (_Float16)a3.z; af[15] = (_Float16)a3.w;

        #pragma unroll
        for (int t = 0; t < 4; ++t) {
            const int ntile = blockIdx.y * 4 + t;
            const v16h bf = Bfv[(ntile * KSTEPS + ks) * 32 + lane];
            acc[t] = __builtin_amdgcn_wmma_f32_16x16x32_f16(
                false, af, false, bf, (short)0, acc[t], false, false);
        }
    }

    #pragma unroll
    for (int t = 0; t < 4; ++t) {
        const int n  = (blockIdx.y * 4 + t) * 16 + lm;
        const float bv = bias ? bias[n] : 0.0f;
        #pragma unroll
        for (int r = 0; r < 8; ++r) {
            const int m = row0 + r + (hi ? 8 : 0);
            float v = acc[t][r] + bv;
            if (do_relu) v = fmaxf(v, 0.0f);
            C[m * ldc + n] = v;
        }
    }
}

// ---------------------------------------------------------------------------
// Pairwise messages: msgs[i,c] = (1/N) * sum_j relu(p[j,c] - p[i,c] + bm[c])
// Block = 16 rows x 64 channels (256 thr); each thread owns 4 rows so every
// LDS read of p[j,c] feeds 4 relu-accumulate chains. p streamed via CDNA5
// GLOBAL_LOAD_ASYNC_TO_LDS_B128 (ASYNCcnt) into a double-buffered 2x16KB LDS
// tile; next chunk in flight while current is consumed. The compute loop
// indexes the __shared__ array DIRECTLY (flattened, runtime base) so reads
// stay addrspace(3) -> ds_load, never flat_load. Writes hu[:, 64:128].
// ---------------------------------------------------------------------------
#define MJ 64   // j-rows per chunk (64*64 floats = 16 KB)
__global__ __launch_bounds__(256)
void msgs_kernel(const float* __restrict__ p,   // [N,64]
                 const float* __restrict__ bm,  // [64]
                 float* __restrict__ hu)        // [N,128]
{
    __shared__ float sp[2 * MJ * 64];
    const int tid = threadIdx.x;
    const int c   = tid & 63;
    const int g   = tid >> 6;                 // 0..3
    const int i0  = blockIdx.x * 16 + g * 4;

    const float pi0 = p[(i0 + 0) * 64 + c];
    const float pi1 = p[(i0 + 1) * 64 + c];
    const float pi2 = p[(i0 + 2) * 64 + c];
    const float pi3 = p[(i0 + 3) * 64 + c];
    const float bmc = bm[c];
    float a0 = 0.0f, a1 = 0.0f, a2 = 0.0f, a3 = 0.0f;

    // Issue first chunk into buf 0: 1024 float4 / 256 threads = 4 per thread.
    {
        const unsigned l0 = lds_off(&sp[0]);
        #pragma unroll
        for (int u = 0; u < 4; ++u) {
            const int idx = tid + u * 256;
            const unsigned laddr = l0 + idx * 16;
            const unsigned long long ga =
                (unsigned long long)(size_t)(p + idx * 4);
            asm volatile("global_load_async_to_lds_b128 %0, %1, off"
                         :: "v"(laddr), "v"(ga) : "memory");
        }
    }

    for (int j0 = 0; j0 < N_NODES; j0 += MJ) {
        const int buf   = (j0 / MJ) & 1;
        const int sbase = buf * (MJ * 64);
        asm volatile("s_wait_asynccnt 0x0" ::: "memory");
        __syncthreads();                      // chunk visible to all waves

        if (j0 + MJ < N_NODES) {              // prefetch next chunk
            const unsigned l1 = lds_off(&sp[(buf ^ 1) * (MJ * 64)]);
            const float* src = p + (j0 + MJ) * 64;
            #pragma unroll
            for (int u = 0; u < 4; ++u) {
                const int idx = tid + u * 256;
                const unsigned laddr = l1 + idx * 16;
                const unsigned long long ga =
                    (unsigned long long)(size_t)(src + idx * 4);
                asm volatile("global_load_async_to_lds_b128 %0, %1, off"
                             :: "v"(laddr), "v"(ga) : "memory");
            }
        }

        #pragma unroll 8
        for (int jj = 0; jj < MJ; ++jj) {
            const float v = sp[sbase + jj * 64 + c] + bmc;   // ds_load_b32
            a0 += fmaxf(v - pi0, 0.0f);
            a1 += fmaxf(v - pi1, 0.0f);
            a2 += fmaxf(v - pi2, 0.0f);
            a3 += fmaxf(v - pi3, 0.0f);
        }
        __syncthreads();                      // done reading buf before reuse
    }

    const float sc = 1.0f / (float)N_NODES;
    hu[(i0 + 0) * 128 + 64 + c] = a0 * sc;
    hu[(i0 + 1) * 128 + 64 + c] = a1 * sc;
    hu[(i0 + 2) * 128 + 64 + c] = a2 * sc;
    hu[(i0 + 3) * 128 + 64 + c] = a3 * sc;
}

// ---------------------------------------------------------------------------
// Heads. Output flat: [0,3072) mean, [3072,3074) std, [3074,4610) value.
// ---------------------------------------------------------------------------
__global__ __launch_bounds__(128)
void heads_kernel(const float* __restrict__ u2,      // [N,128]
                  const float* __restrict__ Wmean,   // [128,2]
                  const float* __restrict__ bmean,   // [2]
                  const float* __restrict__ Wv,      // [128,1]
                  const float* __restrict__ bv,      // [1]
                  const float* __restrict__ log_std, // [2]
                  float* __restrict__ out)
{
    const int i = blockIdx.x * blockDim.x + threadIdx.x;
    if (i < N_NODES) {
        float m0 = 0.0f, m1 = 0.0f, v = 0.0f;
        #pragma unroll 8
        for (int c = 0; c < 128; ++c) {
            const float a = u2[i * 128 + c];
            m0 += a * Wmean[c * 2 + 0];
            m1 += a * Wmean[c * 2 + 1];
            v  += a * Wv[c];
        }
        out[i * 2 + 0] = m0 + bmean[0];
        out[i * 2 + 1] = m1 + bmean[1];
        out[3074 + i]  = v + bv[0];
    }
    if (blockIdx.x == 0 && threadIdx.x < 2)
        out[3072 + threadIdx.x] = expf(log_std[threadIdx.x]);
}

// ---------------------------------------------------------------------------
extern "C" void kernel_launch(void* const* d_in, const int* in_sizes, int n_in,
                              void* d_out, int out_size, void* d_ws, size_t ws_size,
                              hipStream_t stream)
{
    const float* x     = (const float*)d_in[0];
    const float* W1    = (const float*)d_in[1];
    const float* b1    = (const float*)d_in[2];
    const float* W2    = (const float*)d_in[3];
    const float* b2    = (const float*)d_in[4];
    const float* Wm    = (const float*)d_in[5];
    const float* bm    = (const float*)d_in[6];
    const float* Wu1   = (const float*)d_in[7];
    const float* bu1   = (const float*)d_in[8];
    const float* Wu2   = (const float*)d_in[9];
    const float* bu2   = (const float*)d_in[10];
    const float* Wmean = (const float*)d_in[11];
    const float* bmean = (const float*)d_in[12];
    const float* Wv    = (const float*)d_in[13];
    const float* bvv   = (const float*)d_in[14];
    const float* lstd  = (const float*)d_in[15];
    float* out = (float*)d_out;

    // Workspace (floats). u1 reuses h1+p once they're dead. Bf area after u2.
    float* ws = (float*)d_ws;
    float* hu = ws;                           // [1536,128] h | msgs
    float* h1 = ws + 1536 * 128;              // [1536,64]
    float* p  = h1 + 1536 * 64;               // [1536,64]
    float* u1 = ws + 1536 * 128;              // [1536,128] (reuse)
    float* u2 = u1 + 1536 * 128;              // [1536,128]
    _Float16* bf = (_Float16*)(u2 + 1536 * 128);  // swizzled weights (80 KB)
    _Float16* W2f  = bf;                      //  4096 halves
    _Float16* Wmf  = bf + 4096;               //  4096 halves
    _Float16* Wu1f = bf + 8192;               // 16384 halves
    _Float16* Wu2f = bf + 24576;              // 16384 halves

    // one-shot weight swizzles
    swizzle_B_kernel<<<dim3(16), dim3(256), 0, stream>>>(W2,  64,  64,  W2f,  4096);
    swizzle_B_kernel<<<dim3(16), dim3(256), 0, stream>>>(Wm,  64,  64,  Wmf,  4096);
    swizzle_B_kernel<<<dim3(64), dim3(256), 0, stream>>>(Wu1, 128, 128, Wu1f, 16384);
    swizzle_B_kernel<<<dim3(64), dim3(256), 0, stream>>>(Wu2, 128, 128, Wu2f, 16384);

    const dim3 wave(32);

    // h1 = relu(x @ W1 + b1)   (K=4: VALU kernel)
    enc1_kernel<<<dim3(N_NODES * 64 / 256), dim3(256), 0, stream>>>(x, W1, b1, h1);
    // h  = relu(h1 @ W2 + b2) -> hu[:, :64]
    gemm_wmma_kernel<64><<<dim3(96, 1), wave, 0, stream>>>(h1, 64, W2f, b2, hu, 128, 1);
    // p  = h @ Wm  (h = hu cols 0..63, lda=128)
    gemm_wmma_kernel<64><<<dim3(96, 1), wave, 0, stream>>>(hu, 128, Wmf, nullptr, p, 64, 0);
    // msgs -> hu[:, 64:128]  (async-LDS double-buffered pairwise pass)
    msgs_kernel<<<dim3(N_NODES / 16), dim3(256), 0, stream>>>(p, bm, hu);
    // u1 = relu(hu @ Wu1 + bu1)
    gemm_wmma_kernel<128><<<dim3(96, 2), wave, 0, stream>>>(hu, 128, Wu1f, bu1, u1, 128, 1);
    // u2 = relu(u1 @ Wu2 + bu2)
    gemm_wmma_kernel<128><<<dim3(96, 2), wave, 0, stream>>>(u1, 128, Wu2f, bu2, u2, 128, 1);
    // heads -> out
    heads_kernel<<<dim3((N_NODES + 127) / 128), dim3(128), 0, stream>>>(
        u2, Wmean, bmean, Wv, bvv, lstd, out);
}
